// LogSinkhorn_25142738551444
// MI455X (gfx1250) — compile-verified, matrix operand order checked by reading, not run on hardware
//
#include <hip/hip_runtime.h>

typedef _Float16 v4h  __attribute__((ext_vector_type(4)));
typedef _Float16 v8h  __attribute__((ext_vector_type(8)));
typedef _Float16 v16h __attribute__((ext_vector_type(16)));
typedef float    v8f  __attribute__((ext_vector_type(8)));

#define EPS_INV   20.0f     // 1 / 0.05
#define SINK_TOL  0.001f
#define SINK_ITER 50

// ---------------------------------------------------------------------------
// Prep: K = exp(-cost/eps) in f16, plus transposed copy KT (LDS tile transpose)
// grid: (M/32, N/32, B), block 256
// ---------------------------------------------------------------------------
__global__ __launch_bounds__(256)
void sink_prep(const float* __restrict__ cost, _Float16* __restrict__ Kf,
               _Float16* __restrict__ KT, int N, int M) {
  __shared__ _Float16 tile[32][33];
  const int b  = blockIdx.z;
  const int n0 = blockIdx.y * 32;
  const int m0 = blockIdx.x * 32;
  const int t  = threadIdx.x;
  const int r  = t >> 3;          // 0..31
  const int c  = (t & 7) * 4;     // 0,4,...,28
  const size_t NM = (size_t)N * M;

  const float4 cv = *(const float4*)(cost + (size_t)b * NM + (size_t)(n0 + r) * M + (m0 + c));
  v4h h;
  h[0] = (_Float16)__expf(-cv.x * EPS_INV);
  h[1] = (_Float16)__expf(-cv.y * EPS_INV);
  h[2] = (_Float16)__expf(-cv.z * EPS_INV);
  h[3] = (_Float16)__expf(-cv.w * EPS_INV);
  *(v4h*)(Kf + (size_t)b * NM + (size_t)(n0 + r) * M + (m0 + c)) = h;
  tile[r][c + 0] = h[0];
  tile[r][c + 1] = h[1];
  tile[r][c + 2] = h[2];
  tile[r][c + 3] = h[3];
  __syncthreads();

  v4h g;
  g[0] = tile[c + 0][r];
  g[1] = tile[c + 1][r];
  g[2] = tile[c + 2][r];
  g[3] = tile[c + 3][r];
  *(v4h*)(KT + (size_t)b * NM + (size_t)(m0 + r) * N + (n0 + c)) = g;
}

// ---------------------------------------------------------------------------
// Init state: log_u = log_v = 0, log_r/log_c from marginals, flags
// ---------------------------------------------------------------------------
__global__ void sink_init(const float* __restrict__ src, const float* __restrict__ tgt,
                          float* log_u, float* log_v, float* log_r, float* log_c,
                          unsigned* change, int* done, int BN, int BM) {
  const int i = blockIdx.x * blockDim.x + threadIdx.x;
  if (i < BN) { log_u[i] = 0.0f; log_r[i] = __logf(src[i] + 1e-12f); }
  if (i < BM) { log_v[i] = 0.0f; log_c[i] = __logf(tgt[i] + 1e-12f); }
  if (i == 0) { *change = 0u; *done = 0; }
}

// ---------------------------------------------------------------------------
// Per-batch: maxlog = max(logx), expout = exp(logx - maxlog) as f16
// grid: (B), block 256
// ---------------------------------------------------------------------------
__global__ __launch_bounds__(256)
void sink_reduce_exp(const float* __restrict__ logx, int X,
                     float* __restrict__ maxout, _Float16* __restrict__ expout,
                     const int* done) {
  if (*(volatile const int*)done) return;
  const int b = blockIdx.x;
  const int t = threadIdx.x;
  __shared__ float red[256];
  float mx = -3.0e38f;
  for (int i = t; i < X; i += 256) mx = fmaxf(mx, logx[(size_t)b * X + i]);
  red[t] = mx;
  __syncthreads();
  for (int s = 128; s > 0; s >>= 1) {
    if (t < s) red[t] = fmaxf(red[t], red[t + s]);
    __syncthreads();
  }
  const float m = red[0];
  if (t == 0) maxout[b] = m;
  for (int i = t; i < X; i += 256)
    expout[(size_t)b * X + i] = (_Float16)__expf(logx[(size_t)b * X + i] - m);
}

// ---------------------------------------------------------------------------
// WMMA mat-vec + log update:
//   S[r] = sum_c Kmat[b][r][c] * vex[b][c]     (f16 x f16 -> f32 accumulate)
//   state[b][r] = logmarg[b][r] - maxother[b] - log(S[r])
// One 16-row tile per wave, K swept in chunks of 32 with v_wmma_f32_16x16x32_f16.
// B operand = vexp broadcast into all 16 columns (same k-mapping as A operand).
// grid: (R/128, B), block 256 (8 waves)
// ---------------------------------------------------------------------------
__global__ __launch_bounds__(256)
void sink_matvec_wmma(const _Float16* __restrict__ Kmat, const _Float16* __restrict__ vex,
                      const float* __restrict__ maxother, const float* __restrict__ logmarg,
                      float* __restrict__ state, int R, int C,
                      unsigned* __restrict__ change, int* __restrict__ done,
                      int trackChange, int finalize) {
  if (!*(volatile const int*)done) {
    const int b    = blockIdx.y;
    const int wave = threadIdx.x >> 5;
    const int lane = threadIdx.x & 31;
    const int half = lane >> 4;   // 0: k-lo group, 1: k-hi group
    const int lr   = lane & 15;   // row within tile (A) / column (B)
    const int row0 = blockIdx.x * 128 + wave * 16;

    const _Float16* ar = Kmat + ((size_t)b * R + (size_t)(row0 + lr)) * C;
    const _Float16* vr = vex + (size_t)b * C;

    v8f acc = {};
    for (int c0 = 0; c0 < C; c0 += 32) {
      // A tile (16x32 f16): lane lr = row, elements 0-7 -> k = half*8+0..7,
      //                                  elements 8-15 -> k = 16+half*8+0..7
      const v8h a0 = *(const v8h*)(ar + c0 + half * 8);
      const v8h a1 = *(const v8h*)(ar + c0 + 16 + half * 8);
      // B tile (32x16 f16) broadcast: B[k][j] = vex[c0+k] for all j
      const v8h b0 = *(const v8h*)(vr + c0 + half * 8);
      const v8h b1 = *(const v8h*)(vr + c0 + 16 + half * 8);
      const v16h A  = __builtin_shufflevector(a0, a1, 0,1,2,3,4,5,6,7,8,9,10,11,12,13,14,15);
      const v16h Bm = __builtin_shufflevector(b0, b1, 0,1,2,3,4,5,6,7,8,9,10,11,12,13,14,15);
      acc = __builtin_amdgcn_wmma_f32_16x16x32_f16(
          /*neg_a=*/false, A, /*neg_b=*/false, Bm,
          /*c_mod=*/(short)0, acc, /*reuse_a=*/false, /*reuse_b=*/false);
    }

    // D layout: lanes 0-15 hold rows 0..7 in acc[0..7]; lanes 16-31 rows 8..15.
    // Every column j is identical -> take column 0 (lr == 0).
    if (lr == 0) {
      const float mo = maxother[b];
      for (int r = 0; r < 8; ++r) {
        const int   row = row0 + half * 8 + r;
        const float S   = fmaxf(acc[r], 1e-35f);
        const float nw  = logmarg[(size_t)b * R + row] - mo - __logf(S);
        if (trackChange) {
          const float d = fabsf(nw - state[(size_t)b * R + row]);
          atomicMax(change, __float_as_uint(d));   // bits monotone for d >= 0
        }
        state[(size_t)b * R + row] = nw;
      }
    }
  }
  // End-of-iteration convergence update (change was fully written by the
  // preceding row-pass launch; 'done' is only consumed by later launches).
  if (finalize && blockIdx.x == 0 && blockIdx.y == 0 && threadIdx.x == 0) {
    if (!*(volatile const int*)done) {
      const float ch = __uint_as_float(*change);
      if (ch < SINK_TOL) *done = 1;
      *change = 0u;
    }
  }
}

// ---------------------------------------------------------------------------
// Final: T = exp(log_u[n] - cost/eps + log_v[m]), full f32 precision
// ---------------------------------------------------------------------------
__global__ __launch_bounds__(256)
void sink_final(const float* __restrict__ cost, const float* __restrict__ log_u,
                const float* __restrict__ log_v, float* __restrict__ out,
                int N, int M, int B) {
  const size_t i4 = ((size_t)blockIdx.x * blockDim.x + threadIdx.x) * 4;
  const size_t NM = (size_t)N * M;
  if (i4 >= (size_t)B * NM) return;
  const int    b   = (int)(i4 / NM);
  const size_t rem = i4 - (size_t)b * NM;
  const int    n   = (int)(rem / M);
  const int    m   = (int)(rem - (size_t)n * M);
  const float  lu  = log_u[(size_t)b * N + n];
  const float* lv  = log_v + (size_t)b * M + m;
  const float4 cv  = *(const float4*)(cost + i4);
  float4 o;
  o.x = __expf(lu - cv.x * EPS_INV + lv[0]);
  o.y = __expf(lu - cv.y * EPS_INV + lv[1]);
  o.z = __expf(lu - cv.z * EPS_INV + lv[2]);
  o.w = __expf(lu - cv.w * EPS_INV + lv[3]);
  *(float4*)(out + i4) = o;
}

// ---------------------------------------------------------------------------
extern "C" void kernel_launch(void* const* d_in, const int* in_sizes, int n_in,
                              void* d_out, int out_size, void* d_ws, size_t ws_size,
                              hipStream_t stream) {
  const float* cost = (const float*)d_in[0];
  const float* src  = (const float*)d_in[1];
  const float* tgt  = (const float*)d_in[2];

  const size_t BNM = (size_t)in_sizes[0];
  const size_t BN  = (size_t)in_sizes[1];
  const size_t BM  = (size_t)in_sizes[2];
  const int M = (int)(BNM / BN);
  const int N = (int)(BNM / BM);
  const int B = (int)(BN / (size_t)N);

  // Workspace carve (K + K^T stay L2-resident: 2 * 64 MB on 192 MB L2)
  char* w = (char*)d_ws;
  auto carve = [&](size_t bytes) -> char* {
    char* p = w;
    w += (bytes + 255) & ~(size_t)255;
    return p;
  };
  _Float16* Kf    = (_Float16*)carve(BNM * 2);
  _Float16* KT    = (_Float16*)carve(BNM * 2);
  float*    log_u = (float*)carve(BN * 4);
  float*    log_v = (float*)carve(BM * 4);
  float*    log_r = (float*)carve(BN * 4);
  float*    log_c = (float*)carve(BM * 4);
  _Float16* uexp  = (_Float16*)carve(BN * 2);
  _Float16* vexp  = (_Float16*)carve(BM * 2);
  float*    maxu  = (float*)carve((size_t)B * 4);
  float*    maxv  = (float*)carve((size_t)B * 4);
  unsigned* change = (unsigned*)carve(4);
  int*      done   = (int*)carve(4);

  sink_prep<<<dim3(M / 32, N / 32, B), 256, 0, stream>>>(cost, Kf, KT, N, M);
  {
    const int mx = (int)(BN > BM ? BN : BM);
    sink_init<<<(mx + 255) / 256, 256, 0, stream>>>(src, tgt, log_u, log_v, log_r, log_c,
                                                    change, done, (int)BN, (int)BM);
  }
  for (int it = 0; it < SINK_ITER; ++it) {
    // row pass: u update (tracks convergence change)
    sink_reduce_exp<<<B, 256, 0, stream>>>(log_v, M, maxv, vexp, done);
    sink_matvec_wmma<<<dim3(N / 128, B), 256, 0, stream>>>(Kf, vexp, maxv, log_r, log_u,
                                                           N, M, change, done, 1, 0);
    // col pass: v update on transposed copy (same kernel), finalizes iteration
    sink_reduce_exp<<<B, 256, 0, stream>>>(log_u, N, maxu, uexp, done);
    sink_matvec_wmma<<<dim3(M / 128, B), 256, 0, stream>>>(KT, uexp, maxu, log_c, log_v,
                                                           M, N, change, done, 0, 1);
  }
  sink_final<<<(unsigned)((BNM / 4 + 255) / 256), 256, 0, stream>>>(
      cost, log_u, log_v, (float*)d_out, N, M, B);
}
